// TPQwenAttention_17978733101573
// MI455X (gfx1250) — matrix operand order repaired and checked by reading, not compiled, not run on hardware
//
#include <hip/hip_runtime.h>
#include <hip/hip_bf16.h>

#define HIDD 2048
#define NHEAD 16
#define HD 128
#define SEQ 2048
#define BSZ 2

typedef __attribute__((ext_vector_type(16))) __bf16 v16bf;
typedef __attribute__((ext_vector_type(8)))  float  v8f;
typedef __attribute__((ext_vector_type(4)))  unsigned int u32x4;
typedef __attribute__((ext_vector_type(8)))  unsigned int u32x8;

union FragU { uint4 q[2]; v16bf v; };

__device__ __forceinline__ v16bf mkfrag(uint4 a, uint4 b) {
  FragU f; f.q[0] = a; f.q[1] = b; return f.v;
}

__device__ __forceinline__ v8f wmma_bf16(v16bf a, v16bf b, v8f c) {
  return __builtin_amdgcn_wmma_f32_16x16x32_bf16(false, a, false, b, (short)0, c,
                                                 false, false);
}

// CDNA5 async global->LDS copy (ASYNCcnt), 16B per lane, no VGPR round-trip.
__device__ __forceinline__ void async_b128(unsigned int lds_off, const void* g) {
  asm volatile("global_load_async_to_lds_b128 %0, %1, off"
               :: "v"(lds_off), "v"(g) : "memory");
}
__device__ __forceinline__ void wait_async0() {
  asm volatile("s_wait_asynccnt 0" ::: "memory");
}

// ---------------------------------------------------------------- fp32 -> bf16
__global__ void cvt_f32_bf16(const float* __restrict__ in,
                             __hip_bfloat16* __restrict__ out, size_t n) {
  size_t i = ((size_t)blockIdx.x * blockDim.x + threadIdx.x) * 4;
  if (i + 3 < n) {
    float4 f = *(const float4*)(in + i);
    out[i + 0] = __float2bfloat16(f.x);
    out[i + 1] = __float2bfloat16(f.y);
    out[i + 2] = __float2bfloat16(f.z);
    out[i + 3] = __float2bfloat16(f.w);
  } else {
    for (; i < n; ++i) out[i] = __float2bfloat16(in[i]);
  }
}

// ---------------------------------------------------------------- bf16 GEMM
// C[M,N] = A[M,K] * B[K,N] (+ bias). mode 0: bf16 out + bias, mode 1: f32 out.
// 128x64 block tile, 8 waves each 32x32 (2x2 WMMA). BK=32.
// A staged via async-to-LDS into ping-pong buffers; B double-buffered in regs.
#define BM 128
#define BN 64
#define BK 32
#define LDA 40
#define LDB 40

__global__ __launch_bounds__(256) void gemm_bf16(
    const __hip_bfloat16* __restrict__ A, const __hip_bfloat16* __restrict__ Bm,
    const float* __restrict__ bias, void* __restrict__ Cout,
    int M, int N, int K, int mode) {
  __shared__ __hip_bfloat16 sA[2][BM * LDA];   // [row][k]
  __shared__ __hip_bfloat16 sB[2][BN * LDB];   // transposed [n][k]

  const int tid = threadIdx.x;
  const int lane = tid & 31, w = tid >> 5;
  const int lr = lane & 15, hi = lane >> 4;
  const int wm = w >> 1, wn = w & 1;
  const int rw = wm * 32, cw = wn * 32;
  const int m0 = blockIdx.y * BM, n0 = blockIdx.x * BN;

  const int rowA = tid >> 2, segA = tid & 3;   // A chunks: (rowA, rowA+64)
  const int kkB = tid >> 3, nsegB = tid & 7;   // B chunk

  v8f acc[2][2] = {};

  // prologue: stage tile 0
  {
    unsigned int l0 = (unsigned int)(uintptr_t)&sA[0][rowA * LDA + segA * 8];
    async_b128(l0, A + (size_t)(m0 + rowA) * K + segA * 8);
    unsigned int l1 = (unsigned int)(uintptr_t)&sA[0][(rowA + 64) * LDA + segA * 8];
    async_b128(l1, A + (size_t)(m0 + rowA + 64) * K + segA * 8);
    uint4 d = *(const uint4*)(Bm + (size_t)kkB * N + n0 + nsegB * 8);
    const __hip_bfloat16* dh = (const __hip_bfloat16*)&d;
#pragma unroll
    for (int j = 0; j < 8; ++j) sB[0][(nsegB * 8 + j) * LDB + kkB] = dh[j];
  }
  wait_async0();
  __syncthreads();

  const int NK = K / BK;
  for (int ki = 0; ki < NK; ++ki) {
    const int cur = ki & 1, nxt = cur ^ 1;
    const bool more = (ki + 1 < NK);
    const int kn = (ki + 1) * BK;
    uint4 bnext;
    if (more) {
      // async-stage next A tile; issue next B load (consumed after compute)
      unsigned int l0 = (unsigned int)(uintptr_t)&sA[nxt][rowA * LDA + segA * 8];
      async_b128(l0, A + (size_t)(m0 + rowA) * K + kn + segA * 8);
      unsigned int l1 = (unsigned int)(uintptr_t)&sA[nxt][(rowA + 64) * LDA + segA * 8];
      async_b128(l1, A + (size_t)(m0 + rowA + 64) * K + kn + segA * 8);
      bnext = *(const uint4*)(Bm + (size_t)(kn + kkB) * N + n0 + nsegB * 8);
    }

    v16bf af[2], bfr[2];
#pragma unroll
    for (int mt = 0; mt < 2; ++mt) {
      const uint* p = (const uint*)sA[cur] + (rw + mt * 16 + lr) * (LDA / 2) + hi * 4;
      af[mt] = mkfrag(*(const uint4*)p, *(const uint4*)(p + 8));
    }
#pragma unroll
    for (int nt = 0; nt < 2; ++nt) {
      const uint* p = (const uint*)sB[cur] + (cw + nt * 16 + lr) * (LDB / 2) + hi * 4;
      bfr[nt] = mkfrag(*(const uint4*)p, *(const uint4*)(p + 8));
    }
#pragma unroll
    for (int mt = 0; mt < 2; ++mt)
#pragma unroll
      for (int nt = 0; nt < 2; ++nt)
        acc[mt][nt] = wmma_bf16(af[mt], bfr[nt], acc[mt][nt]);

    if (more) {
      const __hip_bfloat16* dh = (const __hip_bfloat16*)&bnext;
#pragma unroll
      for (int j = 0; j < 8; ++j) sB[nxt][(nsegB * 8 + j) * LDB + kkB] = dh[j];
      wait_async0();
    }
    __syncthreads();
  }

#pragma unroll
  for (int mt = 0; mt < 2; ++mt)
#pragma unroll
    for (int nt = 0; nt < 2; ++nt) {
      int row = m0 + rw + mt * 16 + hi * 8;
      int col = n0 + cw + nt * 16 + lr;
      float bv = (mode == 0) ? bias[col] : 0.0f;
#pragma unroll
      for (int r = 0; r < 8; ++r) {
        float v = acc[mt][nt][r] + bv;
        size_t idx = (size_t)(row + r) * N + col;
        if (mode == 0) ((__hip_bfloat16*)Cout)[idx] = __float2bfloat16(v);
        else           ((float*)Cout)[idx] = v;
      }
    }
}

// ---------------------------------------------------------------- attention
// Block = 4 waves handles (b, h, 64 query rows). Flash-style online softmax.
// K block staged by the Tensor Data Mover (one wave issues tensor_load_to_lds,
// padded rows: 64 DWORDs data + 4 DWORDs pad -> 272B LDS stride).
#define QB 64
#define KBL 64
#define LDV 72    // V^T stride (bf16)
#define LDP 72    // P scratch stride (bf16)
#define LDK 136   // TDM-padded K row stride (bf16): 128 + 8

__global__ __launch_bounds__(128) void attn_kernel(
    const __hip_bfloat16* __restrict__ Q, const __hip_bfloat16* __restrict__ Kc,
    const __hip_bfloat16* __restrict__ V, __hip_bfloat16* __restrict__ Ctx) {
  __shared__ __hip_bfloat16 sVt[HD * LDV];       // V^T: [d][key]
  __shared__ __hip_bfloat16 sK[KBL * LDK];       // K:   [key][d] (padded rows)
  __shared__ __hip_bfloat16 sP[4 * 16 * LDP];    // per-wave P scratch

  const int tid = threadIdx.x;
  const int lane = tid & 31, w = tid >> 5;
  const int lr = lane & 15, hi = lane >> 4;
  const int q0 = blockIdx.x * QB;
  const int h = blockIdx.y, b = blockIdx.z;
  const size_t base = ((size_t)b * SEQ) * HIDD + (size_t)h * HD;

  v16bf qf[4];
  {
    const int qrow = q0 + w * 16 + lr;
    const uint* rowp = (const uint*)(Q + base + (size_t)qrow * HIDD);
#pragma unroll
    for (int t = 0; t < 4; ++t) {
      const uint* p = rowp + t * 16 + hi * 4;
      qf[t] = mkfrag(*(const uint4*)p, *(const uint4*)(p + 8));
    }
  }

  const unsigned int ldsK = (unsigned int)(uintptr_t)&sK[0];
  float mrow[8], lrow[8];
  v8f o[8] = {};
#pragma unroll
  for (int r = 0; r < 8; ++r) { mrow[r] = -1e30f; lrow[r] = 0.0f; }
  const float scale = 0.088388347648318447f;  // 1/sqrt(128)
  const int qmax = q0 + QB - 1;

  for (int kb = 0; kb <= qmax; kb += KBL) {
    __syncthreads();  // previous block's reads of sVt/sK complete

    if (w == 0) {
      // TDM descriptor: 2D tile 128 x 64 bf16 from K[kb.., h*128..], padded LDS rows
      unsigned long long ga =
          (unsigned long long)(uintptr_t)(Kc + base + (size_t)kb * HIDD);
      u32x4 g0 = { 1u,                                   // count=1, user mode
                   ldsK,                                 // lds_addr
                   (unsigned int)ga,
                   ((unsigned int)(ga >> 32) & 0x01FFFFFFu) | 0x80000000u }; // type=2
      u32x8 g1 = { (1u << 16) | (1u << 20) | (5u << 22) | (3u << 25),
                   //   data_size=2B | pad_en | pad_interval=64dw | pad_amount=4dw
                   128u << 16,          // tensor_dim0 = 128 (bits 48..79)
                   64u << 16,           // tensor_dim1 = 64  (bits 80..111)
                   128u << 16,          // tile_dim0 = 128   (bits 112..127)
                   64u,                 // tile_dim1 = 64    (bits 128..143)
                   2048u,               // tensor_dim0_stride = HIDD
                   0u, 0u };
      asm volatile("tensor_load_to_lds %0, %1" :: "s"(g0), "s"(g1) : "memory");
    }

    // stage V^T (manual transpose; TDM cannot transpose)
    for (int it = 0; it < 8; ++it) {
      int c = tid + it * 128;
      int key = c >> 4, dseg = c & 15;
      uint4 d4 = *(const uint4*)(V + base + (size_t)(kb + key) * HIDD + dseg * 8);
      const __hip_bfloat16* dh = (const __hip_bfloat16*)&d4;
#pragma unroll
      for (int j = 0; j < 8; ++j) sVt[(dseg * 8 + j) * LDV + key] = dh[j];
    }
    if (w == 0) __builtin_amdgcn_s_wait_tensorcnt(0);
    __syncthreads();

    // S = Q * K^T : K B-fragments now from LDS (TDM-staged, 4x wave reuse)
    float sv[4][8];
#pragma unroll
    for (int nt = 0; nt < 4; ++nt) {
      v8f sa = {};
      const uint* krow = (const uint*)sK + (nt * 16 + lr) * (LDK / 2);
#pragma unroll
      for (int t = 0; t < 4; ++t) {
        const uint* p = krow + t * 16 + hi * 4;
        v16bf kf = mkfrag(*(const uint4*)p, *(const uint4*)(p + 8));
        sa = wmma_bf16(qf[t], kf, sa);
      }
      int kcol = kb + nt * 16 + lr;
#pragma unroll
      for (int r = 0; r < 8; ++r) {
        int qrow = q0 + w * 16 + r + 8 * hi;
        sv[nt][r] = sa[r] * scale + ((kcol > qrow) ? -1e9f : 0.0f);
      }
    }

    // online softmax (rows live in 16-lane half-groups)
    float alpha[8];
#pragma unroll
    for (int r = 0; r < 8; ++r) {
      float mb = fmaxf(fmaxf(sv[0][r], sv[1][r]), fmaxf(sv[2][r], sv[3][r]));
#pragma unroll
      for (int off = 1; off < 16; off <<= 1) mb = fmaxf(mb, __shfl_xor(mb, off, 32));
      float mn = fmaxf(mrow[r], mb);
      alpha[r] = __expf(mrow[r] - mn);
      mrow[r] = mn;
      float ps = 0.0f;
#pragma unroll
      for (int nt = 0; nt < 4; ++nt) { sv[nt][r] = __expf(sv[nt][r] - mn); ps += sv[nt][r]; }
#pragma unroll
      for (int off = 1; off < 16; off <<= 1) ps += __shfl_xor(ps, off, 32);
      lrow[r] = lrow[r] * alpha[r] + ps;
    }
#pragma unroll
    for (int d = 0; d < 8; ++d)
#pragma unroll
      for (int r = 0; r < 8; ++r) o[d][r] *= alpha[r];

    // P: C-layout -> A-fragment layout via LDS bounce
    __hip_bfloat16* pw = sP + w * 16 * LDP;
#pragma unroll
    for (int nt = 0; nt < 4; ++nt)
#pragma unroll
      for (int r = 0; r < 8; ++r)
        pw[(r + 8 * hi) * LDP + nt * 16 + lr] = __float2bfloat16(sv[nt][r]);
    __syncthreads();

    // O += P * V
#pragma unroll
    for (int s = 0; s < 2; ++s) {
      const uint* ap = (const uint*)pw + lr * (LDP / 2) + s * 16 + hi * 4;
      v16bf pa = mkfrag(*(const uint4*)ap, *(const uint4*)(ap + 8));
#pragma unroll
      for (int d = 0; d < 8; ++d) {
        const uint* bp = (const uint*)sVt + (d * 16 + lr) * (LDV / 2) + s * 16 + hi * 4;
        v16bf vb = mkfrag(*(const uint4*)bp, *(const uint4*)(bp + 8));
        o[d] = wmma_bf16(pa, vb, o[d]);
      }
    }
  }

  float inv[8];
#pragma unroll
  for (int r = 0; r < 8; ++r) inv[r] = 1.0f / lrow[r];
#pragma unroll
  for (int d = 0; d < 8; ++d)
#pragma unroll
    for (int r = 0; r < 8; ++r) {
      int qrow = q0 + w * 16 + r + 8 * hi;
      int col = d * 16 + lr;
      Ctx[base + (size_t)qrow * HIDD + col] = __float2bfloat16(o[d][r] * inv[r]);
    }
}

// ---------------------------------------------------------------- launcher
extern "C" void kernel_launch(void* const* d_in, const int* in_sizes, int n_in,
                              void* d_out, int out_size, void* d_ws, size_t ws_size,
                              hipStream_t stream) {
  const float* X  = (const float*)d_in[0];
  // d_in[1] = attention_mask (additive causal) — applied analytically in-kernel
  const float* Wq = (const float*)d_in[2];
  const float* bq = (const float*)d_in[3];
  const float* Wk = (const float*)d_in[4];
  const float* bk = (const float*)d_in[5];
  const float* Wv = (const float*)d_in[6];
  const float* bv = (const float*)d_in[7];
  const float* Wo = (const float*)d_in[8];

  const size_t NX = (size_t)BSZ * SEQ * HIDD;
  const size_t NW = (size_t)HIDD * HIDD;
  __hip_bfloat16* ws  = (__hip_bfloat16*)d_ws;
  __hip_bfloat16* Xb  = ws;
  __hip_bfloat16* Wqb = Xb + NX;
  __hip_bfloat16* Wkb = Wqb + NW;
  __hip_bfloat16* Wvb = Wkb + NW;
  __hip_bfloat16* Wob = Wvb + NW;
  __hip_bfloat16* Qb  = Wob + NW;
  __hip_bfloat16* Kb  = Qb + NX;
  __hip_bfloat16* Vb  = Kb + NX;
  __hip_bfloat16* Cb  = Vb + NX;

  cvt_f32_bf16<<<(int)(NX / 1024), 256, 0, stream>>>(X,  Xb,  NX);
  cvt_f32_bf16<<<(int)(NW / 1024), 256, 0, stream>>>(Wq, Wqb, NW);
  cvt_f32_bf16<<<(int)(NW / 1024), 256, 0, stream>>>(Wk, Wkb, NW);
  cvt_f32_bf16<<<(int)(NW / 1024), 256, 0, stream>>>(Wv, Wvb, NW);
  cvt_f32_bf16<<<(int)(NW / 1024), 256, 0, stream>>>(Wo, Wob, NW);

  const int M = BSZ * SEQ;
  dim3 gg(HIDD / BN, M / BM);
  gemm_bf16<<<gg, 256, 0, stream>>>(Xb, Wqb, bq, Qb, M, HIDD, HIDD, 0);
  gemm_bf16<<<gg, 256, 0, stream>>>(Xb, Wkb, bk, Kb, M, HIDD, HIDD, 0);
  gemm_bf16<<<gg, 256, 0, stream>>>(Xb, Wvb, bv, Vb, M, HIDD, HIDD, 0);

  dim3 ga(SEQ / QB, NHEAD, BSZ);
  attn_kernel<<<ga, 128, 0, stream>>>(Qb, Kb, Vb, Cb);

  gemm_bf16<<<gg, 256, 0, stream>>>(Cb, Wob, nullptr, d_out, M, HIDD, HIDD, 1);
}
